// WeightedInfoNCE_47931835023479
// MI455X (gfx1250) — compile-verified, matrix operand order checked by reading, not run on hardware
//
#include <hip/hip_runtime.h>
#include <math.h>

#define NROWS 8192
#define DDIM  256
#define BM    128
#define BN    128
#define KC    32   // one 16x16x32 K-chunk per LDS stage
#define NCHUNK (DDIM / KC)
#define LB    40   // padded LDS stride in bf16 elements (80B = 5*16B, aligned + bank-spread)

typedef __attribute__((ext_vector_type(8)))  float  v8f;
typedef __attribute__((ext_vector_type(4)))  __bf16 v4bf;
typedef __attribute__((ext_vector_type(8)))  __bf16 v8bf;
typedef __attribute__((ext_vector_type(16))) __bf16 v16bf;
typedef int v4i_vs __attribute__((vector_size(4 * sizeof(int))));

// CDNA5 async global->LDS path (ASYNCcnt) if the toolchain exposes it.
#if defined(__AMDGCN__) && __has_builtin(__builtin_amdgcn_global_load_async_to_lds_b128) && \
    __has_builtin(__builtin_amdgcn_s_wait_asynccnt)
#define WINF_ASYNC 1
#else
#define WINF_ASYNC 0
#endif

__device__ inline void g2l_b128(const __bf16* g, __bf16* l) {
#if WINF_ASYNC
  __builtin_amdgcn_global_load_async_to_lds_b128(
      (v4i_vs*)g,
      (__attribute__((address_space(3))) v4i_vs*)l,
      0, 0);
#else
  *(v8bf*)l = *(const v8bf*)g;
#endif
}

template <int N>
__device__ inline void wait_async_le() {
#if WINF_ASYNC
  __builtin_amdgcn_s_wait_asynccnt(N);
#endif
}

// ---------------------------------------------------------------------------
// Kernel 1: zero the 5 accumulator arrays (rowE, rowS, colE, colS, diag)
// ---------------------------------------------------------------------------
__global__ void winfonce_zero_kernel(float* __restrict__ p, int n) {
  int i = blockIdx.x * blockDim.x + threadIdx.x;
  if (i < n) p[i] = 0.0f;
}

// ---------------------------------------------------------------------------
// Kernel 2a (fallback path): per-row inverse L2 norms
// ---------------------------------------------------------------------------
__global__ void winfonce_rownorm_kernel(const float* __restrict__ f1,
                                        const float* __restrict__ f2,
                                        float* __restrict__ inv1,
                                        float* __restrict__ inv2) {
  const float* src = blockIdx.y ? f2 : f1;
  float*       dst = blockIdx.y ? inv2 : inv1;
  const int wave = threadIdx.x >> 5;
  const int lane = threadIdx.x & 31;
  const int row  = blockIdx.x * 8 + wave;
  const float* p = src + (size_t)row * DDIM;
  float ss = 0.0f;
#pragma unroll
  for (int j = 0; j < DDIM / 32; ++j) {
    float x = p[lane + j * 32];
    ss += x * x;
  }
#pragma unroll
  for (int m = 1; m <= 16; m <<= 1) ss += __shfl_xor(ss, m, 32);
  if (lane == 0) dst[row] = 1.0f / fmaxf(sqrtf(ss), 1e-12f);
}

// ---------------------------------------------------------------------------
// Kernel 2b (main path): normalize rows and split fp32 -> bf16 (hi, lo)
// x_norm ~= h + l with ~17 mantissa bits; wave32 per row, 8 rows per block.
// ---------------------------------------------------------------------------
__global__ void winfonce_normconv_kernel(const float* __restrict__ f1,
                                         const float* __restrict__ f2,
                                         __bf16* __restrict__ Ah, __bf16* __restrict__ Al,
                                         __bf16* __restrict__ Bh, __bf16* __restrict__ Bl) {
  const float* src = blockIdx.y ? f2 : f1;
  __bf16* dh = blockIdx.y ? Bh : Ah;
  __bf16* dl = blockIdx.y ? Bl : Al;
  const int wave = threadIdx.x >> 5;
  const int lane = threadIdx.x & 31;
  const int row  = blockIdx.x * 8 + wave;
  const float* p = src + (size_t)row * DDIM + lane * 8;
  float x[8];
#pragma unroll
  for (int j = 0; j < 8; ++j) x[j] = p[j];
  float ss = 0.0f;
#pragma unroll
  for (int j = 0; j < 8; ++j) ss += x[j] * x[j];
#pragma unroll
  for (int m = 1; m <= 16; m <<= 1) ss += __shfl_xor(ss, m, 32);
  const float inv = 1.0f / fmaxf(sqrtf(ss), 1e-12f);
  v8bf h, l;
#pragma unroll
  for (int j = 0; j < 8; ++j) {
    float v = x[j] * inv;
    __bf16 hb = (__bf16)v;
    h[j] = hb;
    l[j] = (__bf16)(v - (float)hb);
  }
  const size_t o = (size_t)row * DDIM + lane * 8;
  *(v8bf*)&dh[o] = h;
  *(v8bf*)&dl[o] = l;
}

// ---------------------------------------------------------------------------
// Fragment assembly per ISA 16-bit A/B 16x32 layout:
// lane<16: K[ko..ko+7] and K[16+ko..23+ko] with ko=0; lane>=16: ko=8.
// ---------------------------------------------------------------------------
__device__ inline v16bf load_frag(const __bf16* rowPtr, int ko) {
  v8bf r0 = *(const v8bf*)(rowPtr + ko);
  v8bf r1 = *(const v8bf*)(rowPtr + 16 + ko);
  return __builtin_shufflevector(r0, r1, 0, 1, 2, 3, 4, 5, 6, 7,
                                 8, 9, 10, 11, 12, 13, 14, 15);
}

__device__ inline void split4(const float4& x, float s, v4bf& h, v4bf& l) {
  float v0 = x.x * s, v1 = x.y * s, v2 = x.z * s, v3 = x.w * s;
  __bf16 h0 = (__bf16)v0, h1 = (__bf16)v1, h2 = (__bf16)v2, h3 = (__bf16)v3;
  h[0] = h0; h[1] = h1; h[2] = h2; h[3] = h3;
  l[0] = (__bf16)(v0 - (float)h0);
  l[1] = (__bf16)(v1 - (float)h1);
  l[2] = (__bf16)(v2 - (float)h2);
  l[3] = (__bf16)(v3 - (float)h3);
}

// ---------------------------------------------------------------------------
// Kernel 3: fused bf16x3 split-precision WMMA GEMM + InfoNCE statistics.
// 128x128 tile per 256-thread block; wave w owns rows [16w,16w+16).
// sim = (Ah+Al)(Bh+Bl)^T ~= Ah*Bh + Ah*Bl + Al*Bh via v_wmma_f32_16x16x32_bf16.
// PRECONV path: double-buffered LDS + async global->LDS copies (ASYNCcnt).
// ---------------------------------------------------------------------------
template <bool PRECONV>
__global__ void __launch_bounds__(256)
winfonce_simstats_kernel(const float* __restrict__ F1, const float* __restrict__ F2,
                         const __bf16* __restrict__ Ah, const __bf16* __restrict__ Al,
                         const __bf16* __restrict__ Bh, const __bf16* __restrict__ Bl,
                         const float* __restrict__ inv1, const float* __restrict__ inv2,
                         const float* __restrict__ scalePtr,
                         float* __restrict__ rowE, float* __restrict__ rowS,
                         float* __restrict__ colE, float* __restrict__ colS,
                         float* __restrict__ diag) {
  __shared__ __bf16 ldsAh[2][BM * LB];
  __shared__ __bf16 ldsAl[2][BM * LB];
  __shared__ __bf16 ldsBh[2][BN * LB];
  __shared__ __bf16 ldsBl[2][BN * LB];
  __shared__ float colAccE[BN];
  __shared__ float colAccS[BN];

  const int bm   = blockIdx.x;
  const int bn   = blockIdx.y;
  const int tid  = threadIdx.x;
  const int wave = tid >> 5;
  const int lane = tid & 31;
  const int half = lane >> 4;   // 0 or 1
  const int l16  = lane & 15;

  if (tid < BN) { colAccE[tid] = 0.0f; colAccS[tid] = 0.0f; }

  v8f acc[8] = {};

  // 8 async-copy instructions per wave per chunk (4 arrays x 2 iterations)
  auto copyChunk = [&](int b, int kc) {
#pragma unroll
    for (int i = 0; i < 2; ++i) {
      int idx = tid + i * 256;          // 0..511
      int row = idx >> 2;               // 0..127
      int u   = (idx & 3) << 3;         // 0,8,16,24
      size_t gA = (size_t)(bm * BM + row) * DDIM + kc + u;
      size_t gB = (size_t)(bn * BN + row) * DDIM + kc + u;
      g2l_b128(&Ah[gA], &ldsAh[b][row * LB + u]);
      g2l_b128(&Al[gA], &ldsAl[b][row * LB + u]);
      g2l_b128(&Bh[gB], &ldsBh[b][row * LB + u]);
      g2l_b128(&Bl[gB], &ldsBl[b][row * LB + u]);
    }
  };

  auto computeChunk = [&](int b) {
    const int ko = half * 8;
    v16bf a_h = load_frag(&ldsAh[b][(wave * 16 + l16) * LB], ko);
    v16bf a_l = load_frag(&ldsAl[b][(wave * 16 + l16) * LB], ko);
#pragma unroll
    for (int s = 0; s < 8; ++s) {
      v16bf b_h = load_frag(&ldsBh[b][(s * 16 + l16) * LB], ko);
      v16bf b_l = load_frag(&ldsBl[b][(s * 16 + l16) * LB], ko);
      // low-order cross terms first, dominant term last
      acc[s] = __builtin_amdgcn_wmma_f32_16x16x32_bf16(
          false, a_l, false, b_h, (short)0, acc[s], false, false);
      acc[s] = __builtin_amdgcn_wmma_f32_16x16x32_bf16(
          false, a_h, false, b_l, (short)0, acc[s], false, false);
      acc[s] = __builtin_amdgcn_wmma_f32_16x16x32_bf16(
          false, a_h, false, b_h, (short)0, acc[s], false, false);
    }
  };

  if (PRECONV) {
    // ---- 2-stage async pipeline: prefetch chunk i+1 while computing i ----
    copyChunk(0, 0);
#pragma unroll
    for (int i = 0; i < NCHUNK; ++i) {
      const int cur = i & 1;
      if (i + 1 < NCHUNK) {
        copyChunk(1 - cur, (i + 1) * KC);
        // async loads complete in order: <=8 outstanding leaves only chunk i+1
        wait_async_le<8>();
      } else {
        wait_async_le<0>();
      }
      __syncthreads();              // all waves' chunk-i data visible in LDS
      computeChunk(cur);
      __syncthreads();              // done reading 'cur' before it is refilled
    }
  } else {
    // ---- fallback: normalize + split fp32 -> bf16 hi/lo on the fly ----
    for (int kc = 0; kc < DDIM; kc += KC) {
      __syncthreads();
#pragma unroll
      for (int i = 0; i < 4; ++i) {
        int idx = tid + i * 256;          // 0..1023
        int row = idx >> 3;               // 0..127
        int c4  = (idx & 7) << 2;         // 0,4,...,28
        float ia = inv1[bm * BM + row];
        float ib = inv2[bn * BN + row];
        float4 av = *(const float4*)(F1 + (size_t)(bm * BM + row) * DDIM + kc + c4);
        float4 bv = *(const float4*)(F2 + (size_t)(bn * BN + row) * DDIM + kc + c4);
        v4bf ah, al, bh, bl;
        split4(av, ia, ah, al);
        split4(bv, ib, bh, bl);
        *(v4bf*)&ldsAh[0][row * LB + c4] = ah;
        *(v4bf*)&ldsAl[0][row * LB + c4] = al;
        *(v4bf*)&ldsBh[0][row * LB + c4] = bh;
        *(v4bf*)&ldsBl[0][row * LB + c4] = bl;
      }
      __syncthreads();
      computeChunk(0);
    }
  }

  // ---------------- epilogue: scale, exp, fused reductions ----------------
  const float scl = *scalePtr;   // rows already L2-normalized before split
  const int rowBase = bm * BM + wave * 16 + half * 8;   // + r (r=0..7)

  float rowEa[8] = {};
  float rowSa[8] = {};

#pragma unroll
  for (int s = 0; s < 8; ++s) {
    const int gcol = bn * BN + s * 16 + l16;
    float cE = 0.0f, cS = 0.0f;
#pragma unroll
    for (int r = 0; r < 8; ++r) {
      float v = acc[s][r] * scl;          // C/D layout: M=r+8*half, N=l16
      float e = __expf(v);
      rowEa[r] += e; rowSa[r] += v;
      cE += e; cS += v;
      if (rowBase + r == gcol) diag[gcol] = v;   // unique writer
    }
    // fold the two M-halves (lane L and L^16 share a column)
    cE += __shfl_xor(cE, 16, 32);
    cS += __shfl_xor(cS, 16, 32);
    if (half == 0) {
      unsafeAtomicAdd(&colAccE[s * 16 + l16], cE);
      unsafeAtomicAdd(&colAccS[s * 16 + l16], cS);
    }
  }

  // row stats: butterfly over the 16-lane group (sums the 128 columns)
#pragma unroll
  for (int r = 0; r < 8; ++r) {
#pragma unroll
    for (int m = 1; m <= 8; m <<= 1) {
      rowEa[r] += __shfl_xor(rowEa[r], m, 32);
      rowSa[r] += __shfl_xor(rowSa[r], m, 32);
    }
  }
  if (l16 == 0) {  // lanes 0 and 16 hold the two M-half sums
#pragma unroll
    for (int r = 0; r < 8; ++r) {
      unsafeAtomicAdd(&rowE[rowBase + r], rowEa[r]);
      unsafeAtomicAdd(&rowS[rowBase + r], rowSa[r]);
    }
  }

  __syncthreads();
  if (tid < BN) {
    unsafeAtomicAdd(&colE[bn * BN + tid], colAccE[tid]);
    unsafeAtomicAdd(&colS[bn * BN + tid], colAccS[tid]);
  }
}

// ---------------------------------------------------------------------------
// Kernel 4: final scalar reduction.
// per_i = log(rowE_i)+log(colE_i) - 2(1-eps_i)diag_i - eps_i(rowS_i+colS_i)/N
// loss = sum_i per_i / (2N);  eps_i = 1 - 0.9*sigmoid(5 w_i)
// (logits bounded by +-logit_scale, so logsumexp needs no max shift)
// ---------------------------------------------------------------------------
__global__ void __launch_bounds__(1024)
winfonce_loss_kernel(const float* __restrict__ pw,
                     const float* __restrict__ rowE, const float* __restrict__ rowS,
                     const float* __restrict__ colE, const float* __restrict__ colS,
                     const float* __restrict__ diag, float* __restrict__ out) {
  __shared__ float red[32];
  float accum = 0.0f;
  for (int i = threadIdx.x; i < NROWS; i += 1024) {
    float w   = pw[i];
    float sig = 1.0f / (1.0f + __expf(-5.0f * w));
    float eps = 1.0f - 0.9f * sig;
    float per = __logf(rowE[i]) + __logf(colE[i])
              - 2.0f * (1.0f - eps) * diag[i]
              - eps * (rowS[i] + colS[i]) * (1.0f / (float)NROWS);
    accum += per;
  }
#pragma unroll
  for (int m = 1; m <= 16; m <<= 1) accum += __shfl_xor(accum, m, 32);
  const int wave = threadIdx.x >> 5, lane = threadIdx.x & 31;
  if (lane == 0) red[wave] = accum;
  __syncthreads();
  if (wave == 0) {
    float v = red[lane];
#pragma unroll
    for (int m = 1; m <= 16; m <<= 1) v += __shfl_xor(v, m, 32);
    if (lane == 0) out[0] = v / (2.0f * (float)NROWS);
  }
}

// ---------------------------------------------------------------------------
extern "C" void kernel_launch(void* const* d_in, const int* in_sizes, int n_in,
                              void* d_out, int out_size, void* d_ws, size_t ws_size,
                              hipStream_t stream) {
  (void)in_sizes; (void)n_in; (void)out_size;
  const float* f1    = (const float*)d_in[0];
  const float* f2    = (const float*)d_in[1];
  const float* scale = (const float*)d_in[2];
  const float* pw    = (const float*)d_in[3];

  float* ws   = (float*)d_ws;
  float* inv1 = ws + 0 * NROWS;
  float* inv2 = ws + 1 * NROWS;
  float* rowE = ws + 2 * NROWS;
  float* rowS = ws + 3 * NROWS;
  float* colE = ws + 4 * NROWS;
  float* colS = ws + 5 * NROWS;
  float* diag = ws + 6 * NROWS;

  const size_t statsBytes = (size_t)7 * NROWS * sizeof(float);      // 224 KB
  const size_t matBytes   = (size_t)NROWS * DDIM * 2;               // 4 MB each
  __bf16* Ah = (__bf16*)((char*)d_ws + statsBytes);
  __bf16* Al = Ah + (size_t)NROWS * DDIM;
  __bf16* Bh = Al + (size_t)NROWS * DDIM;
  __bf16* Bl = Bh + (size_t)NROWS * DDIM;
  const bool preconv = ws_size >= statsBytes + 4 * matBytes;        // deterministic

  // 1) zero the 5 contiguous accumulator arrays (rowE..diag)
  winfonce_zero_kernel<<<(5 * NROWS + 255) / 256, 256, 0, stream>>>(rowE, 5 * NROWS);

  if (preconv) {
    // 2) normalize + split fp32 -> bf16 hi/lo once
    winfonce_normconv_kernel<<<dim3(NROWS / 8, 2), 256, 0, stream>>>(f1, f2, Ah, Al, Bh, Bl);
    // 3) fused bf16x3 WMMA similarity + statistics (async double-buffered)
    winfonce_simstats_kernel<true><<<dim3(NROWS / BM, NROWS / BN), 256, 0, stream>>>(
        f1, f2, Ah, Al, Bh, Bl, inv1, inv2, scale, rowE, rowS, colE, colS, diag);
  } else {
    winfonce_rownorm_kernel<<<dim3(NROWS / 8, 2), 256, 0, stream>>>(f1, f2, inv1, inv2);
    winfonce_simstats_kernel<false><<<dim3(NROWS / BM, NROWS / BN), 256, 0, stream>>>(
        f1, f2, Ah, Al, Bh, Bl, inv1, inv2, scale, rowE, rowS, colE, colS, diag);
  }

  // 4) final scalar loss
  winfonce_loss_kernel<<<1, 1024, 0, stream>>>(pw, rowE, rowS, colE, colS, diag,
                                               (float*)d_out);
}
